// TemporalAttention_68324339745112
// MI455X (gfx1250) — matrix operand order repaired and checked by reading, not compiled, not run on hardware
//
#include <hip/hip_runtime.h>
#include <hip/hip_bf16.h>

#define DIMSZ   1024
#define NHEADS  16
#define HDIM    64
#define BATCH   2
#define SEQ     2048
#define NTOK    (BATCH * SEQ)   // 4096

typedef __attribute__((ext_vector_type(16))) _Float16 v16h;
typedef __attribute__((ext_vector_type(8)))  float    v8f;
typedef unsigned int u32;
typedef __attribute__((ext_vector_type(4))) u32 u32x4;

union Frag16 {
    v16h  h;
    u32x4 q[2];
    u32   u[8];
    _Float16 e[16];
};

// A-fragment (16xK, 16-bit, K contiguous in memory, row-major, leading dim ld).
// ISA 7.12.2: lanes 0-15 / 16-31 both hold rows M=0..15;
// VGPR v<4 -> K = 2v+{0,1} + 8*half ; v>=4 -> K = 16+2(v-4)+{0,1} + 8*half.
// => dwords {4*half..+3} and {8+4*half..+3}: two b128 loads.
__device__ inline v16h load_a_frag(const _Float16* __restrict__ mat, int row,
                                   int ld, int kbase, int half) {
    Frag16 f;
    const u32* p = (const u32*)(mat + (size_t)row * ld + kbase);
    f.q[0] = *(const u32x4*)(p + 4 * half);
    f.q[1] = *(const u32x4*)(p + 8 + 4 * half);
    return f.h;
}

// B-fragment (Kx16, 16-bit) when memory holds B^T, i.e. (N x K) row-major.
// same K order as A: v<4 -> K = 2v+{0,1} + 8*half ; v>=4 -> K = 16+2(v-4)+{0,1} + 8*half
// => dwords {4*half..+3} and {8+4*half..+3}: two b128 loads.
__device__ inline v16h load_b_frag_nt(const _Float16* __restrict__ mat, int col,
                                      int ld, int kbase, int half) {
    Frag16 f;
    const u32* p = (const u32*)(mat + (size_t)col * ld + kbase);
    f.q[0] = *(const u32x4*)(p + 4 * half);
    f.q[1] = *(const u32x4*)(p + 8 + 4 * half);
    return f.h;
}

// ---------------------------------------------------------------- conversions
__global__ void f32_to_f16_kernel(const float* __restrict__ in,
                                  _Float16* __restrict__ out, int n) {
    const int i = (blockIdx.x * blockDim.x + threadIdx.x) * 4;
    if (i >= n) return;
    const float4 v = *(const float4*)(in + i);
    union { _Float16 h[4]; u32 u[2]; } pk;
    pk.h[0] = (_Float16)v.x; pk.h[1] = (_Float16)v.y;
    pk.h[2] = (_Float16)v.z; pk.h[3] = (_Float16)v.w;
    *(u32*)(out + i)     = pk.u[0];
    *(u32*)(out + i + 2) = pk.u[1];
}

// ---------------------------------------------------------------- QKV GEMM
// qkv = x @ w_qkv^T ; 32x64 tile per wave: 2 A-frags x 4 B-frags -> 8 WMMA/k-step.
// Scatter: Q,K -> [B,H,S,Dh] ; V -> [B,H,Dh,S] (transposed for flash B-frags).
__global__ __launch_bounds__(128)
void qkv_gemm_kernel(const _Float16* __restrict__ x16,
                     const _Float16* __restrict__ w16,
                     _Float16* __restrict__ qbuf,
                     _Float16* __restrict__ kbuf,
                     _Float16* __restrict__ vtbuf) {
    const int wave = threadIdx.x >> 5;
    const int lane = threadIdx.x & 31;
    const int half = lane >> 4;
    const int nidx = lane & 15;
    const int JT = (3 * DIMSZ) / 64;              // 48 col blocks of 64
    const int tile = blockIdx.x * 4 + wave;
    const int tm = tile / JT;
    const int tj = tile % JT;
    const int row0 = tm * 32;
    const int j0   = tj * 64;

    v8f c[2][4] = {};
    for (int k = 0; k < DIMSZ; k += 32) {
        const v16h a0 = load_a_frag(x16, row0 + nidx,      DIMSZ, k, half);
        const v16h a1 = load_a_frag(x16, row0 + 16 + nidx, DIMSZ, k, half);
#pragma unroll
        for (int t = 0; t < 4; ++t) {
            const v16h b = load_b_frag_nt(w16, j0 + 16 * t + nidx, DIMSZ, k, half);
            c[0][t] = __builtin_amdgcn_wmma_f32_16x16x32_f16(false, a0, false, b,
                                                             (short)0, c[0][t], false, false);
            c[1][t] = __builtin_amdgcn_wmma_f32_16x16x32_f16(false, a1, false, b,
                                                             (short)0, c[1][t], false, false);
        }
    }
    // C layout: lane col n fixed; VGPR r -> row m = r + 8*half.
#pragma unroll
    for (int t = 0; t < 4; ++t) {
        const int j   = j0 + 16 * t + nidx;
        const int sel = j >> 10;                  // 0=Q 1=K 2=V
        const int jj  = j & (DIMSZ - 1);
        const int hh  = jj >> 6;
        const int dd  = jj & (HDIM - 1);
#pragma unroll
        for (int i = 0; i < 2; ++i) {
#pragma unroll
            for (int r = 0; r < 8; ++r) {
                const int tok = row0 + 16 * i + r + 8 * half;
                const int bb  = tok >> 11;        // / SEQ
                const int ss  = tok & (SEQ - 1);
                const _Float16 val = (_Float16)c[i][t][r];
                if (sel == 0)
                    qbuf[(((size_t)bb * NHEADS + hh) * SEQ + ss) * HDIM + dd] = val;
                else if (sel == 1)
                    kbuf[(((size_t)bb * NHEADS + hh) * SEQ + ss) * HDIM + dd] = val;
                else
                    vtbuf[(((size_t)bb * NHEADS + hh) * HDIM + dd) * SEQ + ss] = val;
            }
        }
    }
}

// ---------------------------------------------------------------- RoPE (in place)
__global__ void rope_kernel(_Float16* __restrict__ qbuf,
                            _Float16* __restrict__ kbuf) {
    const int total = 2 * BATCH * NHEADS * SEQ * (HDIM / 2);
    int i = blockIdx.x * blockDim.x + threadIdx.x;
    if (i >= total) return;
    const int p = i & 31;  int t = i >> 5;
    const int s = t & (SEQ - 1); t >>= 11;
    const int h = t & (NHEADS - 1); t >>= 4;
    const int b = t & (BATCH - 1);
    const int arr = t >> 1;
    _Float16* buf = arr ? kbuf : qbuf;
    const size_t base = (((size_t)b * NHEADS + h) * SEQ + s) * HDIM;
    const float inv = __powf(10000.0f, -(float)p * (1.0f / 32.0f));
    const float ang = (float)s * inv;
    const float cs = __cosf(ang), sn = __sinf(ang);
    const float lo = (float)buf[base + p];
    const float hi = (float)buf[base + p + 32];
    buf[base + p]      = (_Float16)(lo * cs - hi * sn);
    buf[base + p + 32] = (_Float16)(hi * cs + lo * sn);
}

// ---------------------------------------------------------------- flash attention
// One wave per (b, h, 32-row q block); K/V fragments shared across both 16-row
// q subtiles; online softmax over 32-wide K chunks.
__global__ __launch_bounds__(32)
void flash_attn_kernel(const _Float16* __restrict__ qbuf,
                       const _Float16* __restrict__ kbuf,
                       const _Float16* __restrict__ vtbuf,
                       _Float16* __restrict__ obuf) {      // [B, S, DIM]
    __shared__ __align__(16) _Float16 Pld[32 * 32];
    const int lane = threadIdx.x;
    const int half = lane >> 4;
    const int nidx = lane & 15;
    int blk = blockIdx.x;
    const int qt = blk & (SEQ / 32 - 1); blk >>= 6;
    const int h  = blk & (NHEADS - 1);
    const int b  = blk >> 4;
    const _Float16* Q  = qbuf  + ((size_t)b * NHEADS + h) * SEQ * HDIM;
    const _Float16* Km = kbuf  + ((size_t)b * NHEADS + h) * SEQ * HDIM;
    const _Float16* VT = vtbuf + ((size_t)b * NHEADS + h) * HDIM * SEQ;
    const float scale = 0.125f;                   // 1/sqrt(64)
    const int qrow0 = qt * 32;

    v16h qa[2][2];
#pragma unroll
    for (int i = 0; i < 2; ++i) {
        qa[i][0] = load_a_frag(Q, qrow0 + 16 * i + nidx, HDIM, 0,  half);
        qa[i][1] = load_a_frag(Q, qrow0 + 16 * i + nidx, HDIM, 32, half);
    }

    v8f o[2][4] = {};
    float mrow[2][8], lrow[2][8];
#pragma unroll
    for (int i = 0; i < 2; ++i)
#pragma unroll
        for (int r = 0; r < 8; ++r) { mrow[i][r] = -3.0e38f; lrow[i][r] = 0.0f; }

    const int qend = qrow0 + 31;
    for (int k0 = 0; k0 <= qend; k0 += 32) {
        // ---- scores: S[32x32] = Q(32x64) @ K^T(64x32): 8 WMMA, 4 shared B frags
        v8f s[2][2] = {};
        {
            const v16h kb00 = load_b_frag_nt(Km, k0 + nidx,      HDIM, 0,  half);
            const v16h kb01 = load_b_frag_nt(Km, k0 + nidx,      HDIM, 32, half);
            const v16h kb10 = load_b_frag_nt(Km, k0 + 16 + nidx, HDIM, 0,  half);
            const v16h kb11 = load_b_frag_nt(Km, k0 + 16 + nidx, HDIM, 32, half);
#pragma unroll
            for (int i = 0; i < 2; ++i) {
                s[i][0] = __builtin_amdgcn_wmma_f32_16x16x32_f16(false, qa[i][0], false, kb00, (short)0, s[i][0], false, false);
                s[i][0] = __builtin_amdgcn_wmma_f32_16x16x32_f16(false, qa[i][1], false, kb01, (short)0, s[i][0], false, false);
                s[i][1] = __builtin_amdgcn_wmma_f32_16x16x32_f16(false, qa[i][0], false, kb10, (short)0, s[i][1], false, false);
                s[i][1] = __builtin_amdgcn_wmma_f32_16x16x32_f16(false, qa[i][1], false, kb11, (short)0, s[i][1], false, false);
            }
        }

        // ---- causal mask + online softmax (row = 16*i + r + 8*half)
        float alpha[2][8], rs[2][8];
#pragma unroll
        for (int i = 0; i < 2; ++i) {
#pragma unroll
            for (int r = 0; r < 8; ++r) {
                const int qrow = qrow0 + 16 * i + r + 8 * half;
                float v0 = s[i][0][r] * scale;
                float v1 = s[i][1][r] * scale;
                if (k0 + nidx      > qrow) v0 = -3.0e38f;
                if (k0 + 16 + nidx > qrow) v1 = -3.0e38f;
                float mx = fmaxf(v0, v1);
#pragma unroll
                for (int off = 1; off < 16; off <<= 1)
                    mx = fmaxf(mx, __shfl_xor(mx, off, 32));
                const float mnew = fmaxf(mrow[i][r], mx);
                const float p0 = __expf(v0 - mnew);
                const float p1 = __expf(v1 - mnew);
                float sum = p0 + p1;
#pragma unroll
                for (int off = 1; off < 16; off <<= 1)
                    sum += __shfl_xor(sum, off, 32);
                alpha[i][r] = __expf(mrow[i][r] - mnew);
                rs[i][r]    = sum;
                mrow[i][r]  = mnew;
                const int m = 16 * i + r + 8 * half;
                Pld[m * 32 + nidx]      = (_Float16)p0;   // C-layout -> LDS
                Pld[m * 32 + 16 + nidx] = (_Float16)p1;
            }
        }
        asm volatile("s_wait_dscnt 0" ::: "memory");      // wave-local transpose sync
        __builtin_amdgcn_wave_barrier();

        // ---- P back as A-fragments (2 x 16x32) from LDS
        const v16h pa0 = load_a_frag(Pld, nidx,      32, 0, half);
        const v16h pa1 = load_a_frag(Pld, 16 + nidx, 32, 0, half);

        // ---- rescale running output + stats
#pragma unroll
        for (int i = 0; i < 2; ++i)
#pragma unroll
            for (int r = 0; r < 8; ++r) {
                const float a = alpha[i][r];
                o[i][0][r] *= a; o[i][1][r] *= a; o[i][2][r] *= a; o[i][3][r] *= a;
                lrow[i][r] = lrow[i][r] * a + rs[i][r];
            }

        // ---- O += P(32x32) @ V(32x64): 8 WMMA, 4 shared V frags (V^T contiguous-K)
#pragma unroll
        for (int t = 0; t < 4; ++t) {
            const v16h vb = load_b_frag_nt(VT, 16 * t + nidx, SEQ, k0, half);
            o[0][t] = __builtin_amdgcn_wmma_f32_16x16x32_f16(false, pa0, false, vb, (short)0, o[0][t], false, false);
            o[1][t] = __builtin_amdgcn_wmma_f32_16x16x32_f16(false, pa1, false, vb, (short)0, o[1][t], false, false);
        }

        __builtin_amdgcn_wave_barrier();                  // keep Pld stable vs next iter
    }

    // ---- epilogue: divide by row sum, write [B,S,H*Dh] as f16 for out-proj
#pragma unroll
    for (int i = 0; i < 2; ++i) {
#pragma unroll
        for (int r = 0; r < 8; ++r) {
            const int s = qrow0 + 16 * i + r + 8 * half;
            const float invl = 1.0f / lrow[i][r];
            const size_t base = ((size_t)b * SEQ + s) * DIMSZ + h * HDIM;
            obuf[base + 0 * 16 + nidx] = (_Float16)(o[i][0][r] * invl);
            obuf[base + 1 * 16 + nidx] = (_Float16)(o[i][1][r] * invl);
            obuf[base + 2 * 16 + nidx] = (_Float16)(o[i][2][r] * invl);
            obuf[base + 3 * 16 + nidx] = (_Float16)(o[i][3][r] * invl);
        }
    }
}

// ---------------------------------------------------------------- out projection
// out = attn @ w_out^T ; 32x64 tile per wave, fp32 output.
__global__ __launch_bounds__(128)
void out_gemm_kernel(const _Float16* __restrict__ a16,    // [NTOK, DIM]
                     const _Float16* __restrict__ w16,    // [DIM, DIM]
                     float* __restrict__ out) {           // [NTOK, DIM] fp32
    const int wave = threadIdx.x >> 5;
    const int lane = threadIdx.x & 31;
    const int half = lane >> 4;
    const int nidx = lane & 15;
    const int JT = DIMSZ / 64;                    // 16 col blocks
    const int tile = blockIdx.x * 4 + wave;
    const int tm = tile / JT;
    const int tj = tile % JT;
    const int row0 = tm * 32;
    const int j0   = tj * 64;

    v8f c[2][4] = {};
    for (int k = 0; k < DIMSZ; k += 32) {
        const v16h a0 = load_a_frag(a16, row0 + nidx,      DIMSZ, k, half);
        const v16h a1 = load_a_frag(a16, row0 + 16 + nidx, DIMSZ, k, half);
#pragma unroll
        for (int t = 0; t < 4; ++t) {
            const v16h bfr = load_b_frag_nt(w16, j0 + 16 * t + nidx, DIMSZ, k, half);
            c[0][t] = __builtin_amdgcn_wmma_f32_16x16x32_f16(false, a0, false, bfr,
                                                             (short)0, c[0][t], false, false);
            c[1][t] = __builtin_amdgcn_wmma_f32_16x16x32_f16(false, a1, false, bfr,
                                                             (short)0, c[1][t], false, false);
        }
    }
#pragma unroll
    for (int i = 0; i < 2; ++i)
#pragma unroll
        for (int t = 0; t < 4; ++t)
#pragma unroll
            for (int r = 0; r < 8; ++r)
                out[(size_t)(row0 + 16 * i + r + 8 * half) * DIMSZ + j0 + 16 * t + nidx] = c[i][t][r];
}

// ---------------------------------------------------------------- launch
extern "C" void kernel_launch(void* const* d_in, const int* in_sizes, int n_in,
                              void* d_out, int out_size, void* d_ws, size_t ws_size,
                              hipStream_t stream) {
    (void)in_sizes; (void)n_in; (void)out_size; (void)ws_size;
    const float* x     = (const float*)d_in[0];
    const float* w_qkv = (const float*)d_in[1];
    const float* w_out = (const float*)d_in[2];
    float* out = (float*)d_out;

    char* ws = (char*)d_ws;
    size_t off = 0;
    _Float16* x16    = (_Float16*)(ws + off); off += (size_t)NTOK * DIMSZ * 2;      // 8 MB
    _Float16* wqkv16 = (_Float16*)(ws + off); off += (size_t)3 * DIMSZ * DIMSZ * 2; // 6 MB
    _Float16* wout16 = (_Float16*)(ws + off); off += (size_t)DIMSZ * DIMSZ * 2;     // 2 MB
    _Float16* qb     = (_Float16*)(ws + off); off += (size_t)NTOK * DIMSZ * 2;      // 8 MB
    _Float16* kb     = (_Float16*)(ws + off); off += (size_t)NTOK * DIMSZ * 2;      // 8 MB
    _Float16* vtb    = (_Float16*)(ws + off); off += (size_t)NTOK * DIMSZ * 2;      // 8 MB (transposed)
    _Float16* ab     = (_Float16*)(ws + off); off += (size_t)NTOK * DIMSZ * 2;      // 8 MB

    const int nx = NTOK * DIMSZ, nwq = 3 * DIMSZ * DIMSZ, nwo = DIMSZ * DIMSZ;
    f32_to_f16_kernel<<<(nx / 4  + 255) / 256, 256, 0, stream>>>(x,     x16,    nx);
    f32_to_f16_kernel<<<(nwq / 4 + 255) / 256, 256, 0, stream>>>(w_qkv, wqkv16, nwq);
    f32_to_f16_kernel<<<(nwo / 4 + 255) / 256, 256, 0, stream>>>(w_out, wout16, nwo);

    const int qkv_tiles = (NTOK / 32) * ((3 * DIMSZ) / 64);    // 6144 waves
    qkv_gemm_kernel<<<qkv_tiles / 4, 128, 0, stream>>>(x16, wqkv16, qb, kb, vtb);

    const int rope_threads = 2 * BATCH * NHEADS * SEQ * (HDIM / 2);
    rope_kernel<<<rope_threads / 256, 256, 0, stream>>>(qb, kb);

    flash_attn_kernel<<<BATCH * NHEADS * (SEQ / 32), 32, 0, stream>>>(qb, kb, vtb, ab);

    const int out_tiles = (NTOK / 32) * (DIMSZ / 64);          // 2048 waves
    out_gemm_kernel<<<out_tiles / 4, 128, 0, stream>>>(ab, wout16, out);
}
